// CharacterMaskAttentionHead_64793876627947
// MI455X (gfx1250) — matrix operand order repaired and checked by reading, not compiled, not run on hardware
//
#include <hip/hip_runtime.h>

typedef unsigned short u16;
typedef unsigned int   u32;
typedef unsigned long long u64;

typedef __bf16 v16bf __attribute__((ext_vector_type(16)));
typedef float  v8f   __attribute__((ext_vector_type(8)));
typedef int    v4i   __attribute__((ext_vector_type(4)));

union ABFrag { uint4 q[2]; v16bf v; };

// ---------------- constants ----------------
#define BN   4
#define HH   160
#define WW   160
#define HWF  (HH*WW)      // 25600
#define SG   40
#define SSG  (SG*SG)      // 1600
#define CC   128
#define CIN0 258
#define CINP0 288         // 258 padded to 9*32
#define LL   25
#define KK   16
#define NROW (KK*LL)      // 400 dynamic-kernel rows

// ---------------- gfx1250 async global->LDS path (guarded) ----------------
#if defined(__has_builtin)
#if __has_builtin(__builtin_amdgcn_global_load_async_to_lds_b128) && \
    __has_builtin(__builtin_amdgcn_s_wait_asynccnt)
#define USE_ASYNC_LDS 1
#endif
#endif

#define AS1 __attribute__((address_space(1)))
#define AS3 __attribute__((address_space(3)))

// 16-byte global -> LDS copy (async GLOBAL_LOAD_ASYNC_TO_LDS_B128 on gfx1250)
__device__ __forceinline__ void copy16_g2l(const u16* g, u16* l) {
#ifdef USE_ASYNC_LDS
    __builtin_amdgcn_global_load_async_to_lds_b128(
        (AS1 v4i*)(u64)g,
        (AS3 v4i*)(u32)(u64)l, 0, 0);
#else
    *(uint4*)l = *(const uint4*)g;
#endif
}
__device__ __forceinline__ void async_fence() {
#ifdef USE_ASYNC_LDS
    __builtin_amdgcn_s_wait_asynccnt(0);   // drain ASYNCcnt before the barrier
#endif
}

// ---------------- helpers ----------------
__device__ __forceinline__ u16 f2bf(float f) {
    union { float f; u32 u; } x; x.f = f;
    u32 u = x.u + 0x7FFFu + ((x.u >> 16) & 1u);   // RNE
    return (u16)(u >> 16);
}
__device__ __forceinline__ float bf2f(u16 h) {
    union { u32 u; float f; } x; x.u = ((u32)h) << 16; return x.f;
}

// ---------------- weight repack: OIHW f32 -> [tap][Cout][CinP] bf16 ----------------
__global__ void wconv_kernel(const float* __restrict__ src, u16* __restrict__ dst,
                             int Cin, int CinP, int ntap, int total) {
    int t = blockIdx.x * blockDim.x + threadIdx.x;
    if (t >= total) return;
    int i   = t % CinP;
    int o   = (t / CinP) % CC;
    int tap = t / (CinP * CC);
    float v = (i < Cin) ? src[(size_t)(o * Cin + i) * ntap + tap] : 0.0f;
    dst[t] = f2bf(v);
}

// ---------------- base: concat(feats, xx, yy) NCHW f32 -> NHWC bf16, pad to 288 ----------------
__global__ void build_base_kernel(const float* __restrict__ feats, u16* __restrict__ base) {
    int t = blockIdx.x * blockDim.x + threadIdx.x;
    const int total = BN * HWF * CINP0;
    if (t >= total) return;
    int c = t % CINP0;
    int p = t / CINP0;
    int x = p % WW, y = (p / WW) % HH, b = p / HWF;
    float v;
    if (c < 256)      v = feats[((size_t)(b * 256 + c) * HH + y) * WW + x];
    else if (c == 256) v = -1.0f + 2.0f * (float)x / (float)(WW - 1);
    else if (c == 257) v = -1.0f + 2.0f * (float)y / (float)(HH - 1);
    else               v = 0.0f;
    base[t] = f2bf(v);
}

// ---------------- bilinear resize 160x160 -> 40x40 (NHWC bf16) ----------------
__global__ void resize_kernel(const u16* __restrict__ base, u16* __restrict__ kbase) {
    int t = blockIdx.x * blockDim.x + threadIdx.x;
    const int total = BN * SSG * CINP0;
    if (t >= total) return;
    int c = t % CINP0;
    int g = t / CINP0;
    int gx = g % SG, gy = (g / SG) % SG, b = g / SSG;
    float fx = (gx + 0.5f) * 4.0f - 0.5f;
    float fy = (gy + 0.5f) * 4.0f - 0.5f;
    int x0 = (int)floorf(fx), y0 = (int)floorf(fy);
    float wx = fx - x0, wy = fy - y0;
    int x0c = min(max(x0, 0), WW - 1), x1c = min(max(x0 + 1, 0), WW - 1);
    int y0c = min(max(y0, 0), HH - 1), y1c = min(max(y0 + 1, 0), HH - 1);
    const u16* bp = base + (size_t)b * HWF * CINP0;
    float v00 = bf2f(bp[((size_t)(y0c * WW + x0c)) * CINP0 + c]);
    float v01 = bf2f(bp[((size_t)(y0c * WW + x1c)) * CINP0 + c]);
    float v10 = bf2f(bp[((size_t)(y1c * WW + x0c)) * CINP0 + c]);
    float v11 = bf2f(bp[((size_t)(y1c * WW + x1c)) * CINP0 + c]);
    float v = (v00 * (1 - wx) + v01 * wx) * (1 - wy) + (v10 * (1 - wx) + v11 * wx) * wy;
    kbase[t] = f2bf(v);
}

// ---------------- implicit-GEMM conv (3x3 SAME or 1x1) + fused GN partial stats ----------------
// in   : NHWC bf16 [B*HW, CinP]
// wgt  : bf16 [ntap][Cout=128][CinP]   (B-matrix staged in LDS as [n][k])
// raw  : NHWC f32  [B*HW, 128]
// stats: f32 [B][32][2] (sum, sumsq) via atomics — caller zeroes first
// grid: (Mtotal/MT, 128/64), block 256 (8 waves: 4 in M x 2 in N)
// MT must divide HW (keeps per-block batch uniform for the fused GN stats)
template<int MT>
__global__ __launch_bounds__(256)
void conv_wmma_kernel(const u16* __restrict__ in, const u16* __restrict__ wgt,
                      float* __restrict__ raw, float* __restrict__ stats,
                      int Hin, int Win, int HW, int CinP, int k3) {
    constexpr int MSUB = MT / 64;            // 16-row M-subtiles per wave
    constexpr int U    = MT / 64;            // uint4 A-stage transfers per thread
    const int ntap  = k3 ? 9 : 1;
    const int mBase = blockIdx.x * MT;
    const int nBase = blockIdx.y * 64;
    const int tid  = threadIdx.x;
    const int wave = tid >> 5, lane = tid & 31;
    const int wm = wave & 3, wn = wave >> 2;
    const int b = mBase / HW;

    __shared__ u16 ldsA[MT * 32];   // [m][k]
    __shared__ u16 ldsB[64 * 32];   // [n][k]

    const v8f vz = {0, 0, 0, 0, 0, 0, 0, 0};
    v8f acc[2][MSUB];
    for (int h = 0; h < 2; ++h)
        for (int ms = 0; ms < MSUB; ++ms) acc[h][ms] = vz;

    // A staging: MT rows x 4 segs of 8 bf16; each thread owns one row, U segs
    const int arow  = (U == 1) ? (tid >> 2) : (tid >> 1);
    const int aseg0 = (U == 1) ? (tid & 3) : ((tid & 1) * 2);
    const int pixr = mBase + arow - b * HW;
    const int py = pixr / Win, px = pixr % Win;
    // B staging: 64 n x 4 segs of 8 bf16
    const int brow = tid >> 2, bseg = tid & 3;

    for (int t = 0; t < ntap; ++t) {
        const int ty = k3 ? (t / 3 - 1) : 0;
        const int tx = k3 ? (t % 3 - 1) : 0;
        const int sy = py + ty, sx = px + tx;
        const bool inb = (sy >= 0) & (sy < Hin) & (sx >= 0) & (sx < Win);
        const u16* arowp = in + (size_t)(b * HW + sy * Win + sx) * CinP;
        const u16* wtap  = wgt + (size_t)t * CC * CinP;

        for (int kc = 0; kc < CinP; kc += 32) {
            __syncthreads();
            for (int i = 0; i < U; ++i) {
                const int seg = aseg0 + i;
                u16* lp = &ldsA[arow * 32 + seg * 8];
                if (inb) {
                    copy16_g2l(arowp + kc + seg * 8, lp);
                } else {
                    uint4 z = {0u, 0u, 0u, 0u};
                    *(uint4*)lp = z;                 // DScnt store, covered by barrier
                }
            }
            copy16_g2l(wtap + (size_t)(nBase + brow) * CinP + kc + bseg * 8,
                       &ldsB[brow * 32 + bseg * 8]);
            async_fence();
            __syncthreads();

            // B fragments (32x16): lane<16 -> N=lane, K={0..7,16..23}; lane>=16 -> K={8..15,24..31}
            ABFrag bfr[2];
            const int bn = wn * 32 + (lane & 15);
            const int bk = (lane < 16) ? 0 : 8;
            bfr[0].q[0] = *(const uint4*)(&ldsB[bn * 32 + bk]);
            bfr[0].q[1] = *(const uint4*)(&ldsB[bn * 32 + bk + 16]);
            bfr[1].q[0] = *(const uint4*)(&ldsB[(bn + 16) * 32 + bk]);
            bfr[1].q[1] = *(const uint4*)(&ldsB[(bn + 16) * 32 + bk + 16]);

            // A fragments (16x32): lane<16 -> M=lane, K={0..7,16..23}; lane>=16 -> K={8..15,24..31}
            const int ak = (lane < 16) ? 0 : 8;
            for (int ms = 0; ms < MSUB; ++ms) {
                ABFrag af;
                const int am = wm * (16 * MSUB) + ms * 16 + (lane & 15);
                af.q[0] = *(const uint4*)(&ldsA[am * 32 + ak]);
                af.q[1] = *(const uint4*)(&ldsA[am * 32 + ak + 16]);
                acc[0][ms] = __builtin_amdgcn_wmma_f32_16x16x32_bf16(
                    false, af.v, false, bfr[0].v, (short)0, acc[0][ms], false, false);
                acc[1][ms] = __builtin_amdgcn_wmma_f32_16x16x32_bf16(
                    false, af.v, false, bfr[1].v, (short)0, acc[1][ms], false, false);
            }
        }
    }

    // D layout: VGPR j, lanes 0-15 -> M=j, N=lane ; lanes 16-31 -> M=j+8, N=lane-16
    for (int h = 0; h < 2; ++h) {
        const int n = nBase + wn * 32 + h * 16 + (lane & 15);
        float s = 0.0f, q = 0.0f;
        for (int ms = 0; ms < MSUB; ++ms) {
            const int mLane = mBase + wm * (16 * MSUB) + ms * 16 + ((lane < 16) ? 0 : 8);
            for (int j = 0; j < 8; ++j) {
                float v = acc[h][ms][j];
                raw[(size_t)(mLane + j) * CC + n] = v;
                s += v; q += v * v;
            }
        }
        const int g = n >> 2;                   // 4 channels per GN group
        atomicAdd(&stats[(b * 32 + g) * 2 + 0], s);
        atomicAdd(&stats[(b * 32 + g) * 2 + 1], q);
    }
}

// ---------------- GN finalize: stats -> per (b,c) scale/shift ----------------
__global__ void gn_finalize_kernel(const float* __restrict__ stats,
                                   const float* __restrict__ gamma,
                                   const float* __restrict__ beta,
                                   float* __restrict__ scsh, int HW) {
    int t = blockIdx.x * blockDim.x + threadIdx.x;
    if (t >= BN * CC) return;
    int b = t / CC, c = t % CC, g = c >> 2;
    float cnt = 4.0f * (float)HW;
    float s = stats[(b * 32 + g) * 2 + 0];
    float q = stats[(b * 32 + g) * 2 + 1];
    float mean = s / cnt;
    float var = q / cnt - mean * mean;
    float sc = gamma[c] * rsqrtf(var + 1e-5f);
    scsh[t * 2 + 0] = sc;
    scsh[t * 2 + 1] = beta[c] - mean * sc;
}

// ---------------- apply GN + ReLU, f32 -> bf16 NHWC ----------------
__global__ void apply_kernel(const float* __restrict__ raw, const float* __restrict__ scsh,
                             u16* __restrict__ out, int total, int HW) {
    int t = blockIdx.x * blockDim.x + threadIdx.x;
    if (t >= total) return;
    int c = t % CC;
    int b = (t / CC) / HW;
    float sc = scsh[(b * CC + c) * 2 + 0];
    float sh = scsh[(b * CC + c) * 2 + 1];
    float v = raw[t] * sc + sh;
    out[t] = f2bf(v > 0.0f ? v : 0.0f);
}

// ---------------- stable top-k (values in [0,8), K=16) ----------------
__global__ void topk_kernel(const int* __restrict__ mb, int* __restrict__ topk) {
    int b = threadIdx.x;
    if (b >= BN) return;
    const int* v = mb + b * SSG;
    int cnt = 0;
    for (int val = 7; val >= 0 && cnt < KK; --val)
        for (int i = 0; i < SSG && cnt < KK; ++i)
            if (v[i] == val) topk[b * KK + cnt++] = i;
}

// ---------------- gather top-k cells + 1x1 kow projection -> kern bf16 [B][400][128] ----------------
__global__ void kern_build_kernel(const u16* __restrict__ kfeat, const float* __restrict__ kow,
                                  const float* __restrict__ kob, const int* __restrict__ topk,
                                  u16* __restrict__ kern) {
    int t = blockIdx.x * blockDim.x + threadIdx.x;
    const int total = BN * CC * LL * KK;    // 204800
    if (t >= total) return;
    int k = t % KK;
    int o = (t / KK) % (CC * LL);           // kow output channel = l*128 + c
    int b = t / (KK * CC * LL);
    int cell = topk[b * KK + k];
    const u16*   fp = kfeat + (size_t)(b * SSG + cell) * CC;
    const float* wp = kow + (size_t)o * CC;
    float acc = kob[o];
    for (int ci = 0; ci < CC; ++ci) acc += bf2f(fp[ci]) * wp[ci];
    int l = o >> 7, c = o & 127;
    kern[((size_t)(b * NROW + k * LL + l)) * CC + c] = f2bf(acc);
}

// ---------------- dynamic conv: out[b, kl, p] = sum_c kern[b,kl,c] * mf[b,p,c] ----------------
// M = pixels (contiguous NHWC loads), N = 16 kernel rows, K = 128
// grid: (B * HW/128, 25), block 256 (8 waves along M)
__global__ __launch_bounds__(256)
void dyn_conv_kernel(const u16* __restrict__ mf, const u16* __restrict__ kern,
                     float* __restrict__ out) {
    const int bx = blockIdx.x;
    const int b = bx / (HWF / 128);
    const int pBase = (bx % (HWF / 128)) * 128;
    const int nBase = blockIdx.y * 16;
    const int tid = threadIdx.x;
    const int wave = tid >> 5, lane = tid & 31;

    __shared__ u16 ldsA[128 * 32];  // [m][k]
    __shared__ u16 ldsB[16 * 32];   // [n][k]

    v8f acc = {0, 0, 0, 0, 0, 0, 0, 0};

    for (int kc = 0; kc < CC; kc += 32) {
        __syncthreads();
        const int ar = tid >> 1, as = tid & 1;
        const u16* ap = mf + ((size_t)(b * HWF + pBase + ar)) * CC + kc + as * 16;
        copy16_g2l(ap + 0, &ldsA[ar * 32 + as * 16 + 0]);
        copy16_g2l(ap + 8, &ldsA[ar * 32 + as * 16 + 8]);
        if (tid < 64) {
            const int bn = tid >> 2, bs = tid & 3;
            copy16_g2l(kern + ((size_t)(b * NROW + nBase + bn)) * CC + kc + bs * 8,
                       &ldsB[bn * 32 + bs * 8]);
        }
        async_fence();
        __syncthreads();

        ABFrag af, bfr;
        const int am = wave * 16 + (lane & 15);
        const int ak = (lane < 16) ? 0 : 8;
        af.q[0] = *(const uint4*)(&ldsA[am * 32 + ak]);
        af.q[1] = *(const uint4*)(&ldsA[am * 32 + ak + 16]);
        const int bn = lane & 15;
        const int bk = (lane < 16) ? 0 : 8;
        bfr.q[0] = *(const uint4*)(&ldsB[bn * 32 + bk]);
        bfr.q[1] = *(const uint4*)(&ldsB[bn * 32 + bk + 16]);

        acc = __builtin_amdgcn_wmma_f32_16x16x32_bf16(false, af.v, false, bfr.v,
                                                      (short)0, acc, false, false);
    }

    const int n = nBase + (lane & 15);
    const int mOff = wave * 16 + ((lane < 16) ? 0 : 8);
    float* op = out + ((size_t)(b * NROW + n)) * HWF + pBase + mOff;
    float4 lo = {acc[0], acc[1], acc[2], acc[3]};
    float4 hi = {acc[4], acc[5], acc[6], acc[7]};
    *(float4*)(op + 0) = lo;
    *(float4*)(op + 4) = hi;
}

// ==================================================================================
extern "C" void kernel_launch(void* const* d_in, const int* in_sizes, int n_in,
                              void* d_out, int out_size, void* d_ws, size_t ws_size,
                              hipStream_t stream) {
    (void)in_sizes; (void)n_in; (void)out_size; (void)ws_size;
    char* ws = (char*)d_ws;

    // ---- inputs ----
    const float* feats = (const float*)d_in[0];
    const int*   mbox  = (const int*)d_in[1];
    const float* kw[4] = { (const float*)d_in[2],  (const float*)d_in[5],
                           (const float*)d_in[8],  (const float*)d_in[11] };
    const float* kg[4] = { (const float*)d_in[3],  (const float*)d_in[6],
                           (const float*)d_in[9],  (const float*)d_in[12] };
    const float* kb[4] = { (const float*)d_in[4],  (const float*)d_in[7],
                           (const float*)d_in[10], (const float*)d_in[13] };
    const float* fw[4] = { (const float*)d_in[14], (const float*)d_in[17],
                           (const float*)d_in[20], (const float*)d_in[23] };
    const float* fg[4] = { (const float*)d_in[15], (const float*)d_in[18],
                           (const float*)d_in[21], (const float*)d_in[24] };
    const float* fb[4] = { (const float*)d_in[16], (const float*)d_in[19],
                           (const float*)d_in[22], (const float*)d_in[25] };
    const float* kow = (const float*)d_in[26];
    const float* kob = (const float*)d_in[27];
    const float* fow = (const float*)d_in[28];
    const float* fog = (const float*)d_in[29];
    const float* fob = (const float*)d_in[30];
    float* out = (float*)d_out;

    // ---- workspace layout ----
    size_t off = 0;
    auto take = [&](size_t bytes) { size_t o = off; off = (off + bytes + 255) & ~(size_t)255; return o; };
    const size_t oBASE  = take((size_t)BN * HWF * CINP0 * 2);
    const size_t oKBASE = take((size_t)BN * SSG * CINP0 * 2);
    const size_t oACTA  = take((size_t)BN * HWF * CC * 2);
    const size_t oACTB  = take((size_t)BN * HWF * CC * 2);
    const size_t oRAW   = take((size_t)BN * HWF * CC * 4);
    const size_t oMF    = take((size_t)BN * HWF * CC * 2);
    const size_t oKFEAT = take((size_t)BN * SSG * CC * 2);
    const size_t oKERN  = take((size_t)BN * NROW * CC * 2);
    const size_t oSTATS = take((size_t)BN * 32 * 2 * 4);
    const size_t oSCSH  = take((size_t)BN * CC * 2 * 4);
    const size_t oTOPK  = take((size_t)BN * KK * 4);
    size_t oWF[4], oWK[4];
    oWF[0] = take((size_t)9 * CC * CINP0 * 2);
    for (int l = 1; l < 4; ++l) oWF[l] = take((size_t)9 * CC * CC * 2);
    oWK[0] = take((size_t)9 * CC * CINP0 * 2);
    for (int l = 1; l < 4; ++l) oWK[l] = take((size_t)9 * CC * CC * 2);
    const size_t oWFO = take((size_t)CC * CC * 2);

    u16*   BASE  = (u16*)(ws + oBASE);
    u16*   KBASE = (u16*)(ws + oKBASE);
    u16*   ACTA  = (u16*)(ws + oACTA);
    u16*   ACTB  = (u16*)(ws + oACTB);
    float* RAW   = (float*)(ws + oRAW);
    u16*   MF    = (u16*)(ws + oMF);
    u16*   KFEAT = (u16*)(ws + oKFEAT);
    u16*   KERN  = (u16*)(ws + oKERN);
    float* STATS = (float*)(ws + oSTATS);
    float* SCSH  = (float*)(ws + oSCSH);
    int*   TOPK  = (int*)(ws + oTOPK);

    const int TB = 256;
    auto blocks = [](long long n, int tb) { return (unsigned)((n + tb - 1) / tb); };

    // ---- weight repack ----
    {
        long long n0 = (long long)9 * CC * CINP0;
        wconv_kernel<<<blocks(n0, TB), TB, 0, stream>>>(fw[0], (u16*)(ws + oWF[0]), CIN0, CINP0, 9, (int)n0);
        wconv_kernel<<<blocks(n0, TB), TB, 0, stream>>>(kw[0], (u16*)(ws + oWK[0]), CIN0, CINP0, 9, (int)n0);
        long long n1 = (long long)9 * CC * CC;
        for (int l = 1; l < 4; ++l) {
            wconv_kernel<<<blocks(n1, TB), TB, 0, stream>>>(fw[l], (u16*)(ws + oWF[l]), CC, CC, 9, (int)n1);
            wconv_kernel<<<blocks(n1, TB), TB, 0, stream>>>(kw[l], (u16*)(ws + oWK[l]), CC, CC, 9, (int)n1);
        }
        long long n2 = (long long)CC * CC;
        wconv_kernel<<<blocks(n2, TB), TB, 0, stream>>>(fow, (u16*)(ws + oWFO), CC, CC, 1, (int)n2);
    }

    // ---- base + resize ----
    build_base_kernel<<<blocks((long long)BN * HWF * CINP0, TB), TB, 0, stream>>>(feats, BASE);
    resize_kernel<<<blocks((long long)BN * SSG * CINP0, TB), TB, 0, stream>>>(BASE, KBASE);

    // feature branch: 128-row tiles (25600 % 128 == 0)
    auto cgrBig = [&](const u16* in, size_t wOff, const float* g, const float* be,
                      u16* act, int CinP, int k3) {
        (void)hipMemsetAsync(STATS, 0, (size_t)BN * 32 * 2 * 4, stream);
        dim3 grid((unsigned)(BN * HWF / 128), 2);
        conv_wmma_kernel<128><<<grid, 256, 0, stream>>>(in, (const u16*)(ws + wOff), RAW, STATS,
                                                        HH, WW, HWF, CinP, k3);
        gn_finalize_kernel<<<2, 256, 0, stream>>>(STATS, g, be, SCSH, HWF);
        long long tot = (long long)BN * HWF * CC;
        apply_kernel<<<blocks(tot, TB), TB, 0, stream>>>(RAW, SCSH, act, (int)tot, HWF);
    };
    // kernel branch: 64-row tiles (1600 % 64 == 0)
    auto cgrSmall = [&](const u16* in, size_t wOff, const float* g, const float* be,
                        u16* act, int CinP) {
        (void)hipMemsetAsync(STATS, 0, (size_t)BN * 32 * 2 * 4, stream);
        dim3 grid((unsigned)(BN * SSG / 64), 2);
        conv_wmma_kernel<64><<<grid, 256, 0, stream>>>(in, (const u16*)(ws + wOff), RAW, STATS,
                                                       SG, SG, SSG, CinP, 1);
        gn_finalize_kernel<<<2, 256, 0, stream>>>(STATS, g, be, SCSH, SSG);
        long long tot = (long long)BN * SSG * CC;
        apply_kernel<<<blocks(tot, TB), TB, 0, stream>>>(RAW, SCSH, act, (int)tot, SSG);
    };

    // ---- feature branch: 4x (3x3 conv + GN + ReLU) + 1x1 conv + GN + ReLU ----
    cgrBig(BASE, oWF[0], fg[0], fb[0], ACTA, CINP0, 1);
    cgrBig(ACTA, oWF[1], fg[1], fb[1], ACTB, CC, 1);
    cgrBig(ACTB, oWF[2], fg[2], fb[2], ACTA, CC, 1);
    cgrBig(ACTA, oWF[3], fg[3], fb[3], ACTB, CC, 1);
    cgrBig(ACTB, oWFO,   fog,   fob,   MF,   CC, 0);

    // ---- kernel branch: 4x (3x3 conv + GN + ReLU) on 40x40 grid ----
    cgrSmall(KBASE, oWK[0], kg[0], kb[0], ACTA, CINP0);
    cgrSmall(ACTA,  oWK[1], kg[1], kb[1], ACTB, CC);
    cgrSmall(ACTB,  oWK[2], kg[2], kb[2], ACTA, CC);
    cgrSmall(ACTA,  oWK[3], kg[3], kb[3], KFEAT, CC);

    // ---- top-k select + dynamic kernel build (gather-then-project) ----
    topk_kernel<<<1, BN, 0, stream>>>(mbox, TOPK);
    kern_build_kernel<<<blocks((long long)BN * CC * LL * KK, TB), TB, 0, stream>>>(
        KFEAT, kow, kob, TOPK, KERN);

    // ---- dynamic 1x1 conv via WMMA ----
    {
        dim3 grid((unsigned)(BN * (HWF / 128)), LL);
        dyn_conv_kernel<<<grid, 256, 0, stream>>>(MF, KERN, out);
    }
}
